// QuantLlamaMLP_62251255988371
// MI455X (gfx1250) — 8-bit matrix operand order repaired and checked by reading, compiled, not run on hardware
//
#include <hip/hip_runtime.h>
#include <cstdint>
#include <cstddef>

#define N_TOK 2048
#define HID   4096
#define IMED  11008
#define BK    64
#define LDS_STRIDE 80

typedef __attribute__((ext_vector_type(8))) int v8i;


__device__ __forceinline__ uint32_t lds_off32(const void* p) {
  return (uint32_t)(uintptr_t)p;
}

__device__ __forceinline__ void async_b128(uint32_t lds, const void* base, int voff) {
  asm volatile("global_load_async_to_lds_b128 %0, %1, %2"
               :: "v"(lds), "v"(voff), "s"(base)
               : "memory");
}

__device__ __forceinline__ v8i frag_a(const char* tile, int m0, int lane) {
  const char* p = tile + (size_t)(m0 + (lane & 15)) * LDS_STRIDE + ((lane >> 4) << 4);
  int4 d0 = *(const int4*)(p +  0);
  int4 d2 = *(const int4*)(p + 32);
  v8i f;
  f[0] = d0.x; f[1] = d0.y; f[2] = d0.z; f[3] = d0.w;
  f[4] = d2.x; f[5] = d2.y; f[6] = d2.z; f[7] = d2.w;
  return f;
}

__device__ __forceinline__ v8i frag_b(const char* tile, int n0, int lane) {
  const char* p = tile + (size_t)(n0 + (lane & 15)) * LDS_STRIDE + ((lane >> 4) << 4);
  int4 lo = *(const int4*)(p +  0);
  int4 hi = *(const int4*)(p + 32);
  v8i f;
  f[0] = lo.x; f[1] = lo.y; f[2] = lo.z; f[3] = lo.w;
  f[4] = hi.x; f[5] = hi.y; f[6] = hi.z; f[7] = hi.w;
  return f;
}

__device__ __forceinline__ int q8clamp(float v) {
  int q = (int)rintf(v);
  q = q < -127 ? -127 : q;
  q = q >  127 ?  127 : q;
  return q & 0xff;
}

__global__ __launch_bounds__(256)
void pack_i32_to_i8(const int4* __restrict__ in, int* __restrict__ out, int n4) {
  int i = blockIdx.x * 256 + threadIdx.x;
  if (i < n4) {
    int4 v = in[i];
    out[i] = (v.x & 0xff) | ((v.y & 0xff) << 8) | ((v.z & 0xff) << 16) | ((v.w & 0xff) << 24);
  }
}

__global__ __launch_bounds__(256)
void gateup_silu_kernel(const char* __restrict__ x8,
                        const char* __restrict__ wg8,
                        const float* __restrict__ xscale,
                        const float* __restrict__ wscale,
                        float* __restrict__ act,
                        unsigned* __restrict__ s2bits) {
  __shared__ char sA[2][64  * LDS_STRIDE];
  __shared__ char sG[2][128 * LDS_STRIDE];
  __shared__ char sU[2][128 * LDS_STRIDE];

  const int tid  = threadIdx.x;
  const int lane = tid & 31;
  const int wave = tid >> 5;
  const int wm   = wave & 1;
  const int wn   = wave >> 1;
  const int bm   = blockIdx.x * 64;
  const int bn   = blockIdx.y * 128;

  v8i accG[2][2] = {};
  v8i accU[2][2] = {};

  auto stage = [&](int k0, int b) {
    {
      int row = tid >> 2, ch = (tid & 3) << 4;
      async_b128(lds_off32(&sA[b][row * LDS_STRIDE + ch]), x8,
                 (bm + row) * HID + k0 + ch);
    }
    #pragma unroll
    for (int s = 0; s < 2; ++s) {
      int t = tid + (s << 8);
      int row = t >> 2, ch = (t & 3) << 4;
      async_b128(lds_off32(&sG[b][row * LDS_STRIDE + ch]), wg8,
                 (bn + row) * HID + k0 + ch);
      async_b128(lds_off32(&sU[b][row * LDS_STRIDE + ch]), wg8,
                 (IMED + bn + row) * HID + k0 + ch);
    }
  };

  stage(0, 0);
  for (int k0 = 0, it = 0; k0 < HID; k0 += BK, ++it) {
    const int b = it & 1;
    if (k0 + BK < HID) {
      stage(k0 + BK, b ^ 1);
      asm volatile("s_wait_asynccnt 5" ::: "memory");
    } else {
      asm volatile("s_wait_asynccnt 0" ::: "memory");
    }
    __syncthreads();

    v8i aF0 = frag_a(sA[b], wm * 32 +  0, lane);
    v8i aF1 = frag_a(sA[b], wm * 32 + 16, lane);
    v8i gF0 = frag_b(sG[b], wn * 32 +  0, lane);
    v8i gF1 = frag_b(sG[b], wn * 32 + 16, lane);
    v8i uF0 = frag_b(sU[b], wn * 32 +  0, lane);
    v8i uF1 = frag_b(sU[b], wn * 32 + 16, lane);

    accG[0][0] = __builtin_amdgcn_wmma_i32_16x16x64_iu8(true, aF0, true, gF0, accG[0][0], false, false);
    accG[0][1] = __builtin_amdgcn_wmma_i32_16x16x64_iu8(true, aF0, true, gF1, accG[0][1], false, false);
    accG[1][0] = __builtin_amdgcn_wmma_i32_16x16x64_iu8(true, aF1, true, gF0, accG[1][0], false, false);
    accG[1][1] = __builtin_amdgcn_wmma_i32_16x16x64_iu8(true, aF1, true, gF1, accG[1][1], false, false);
    accU[0][0] = __builtin_amdgcn_wmma_i32_16x16x64_iu8(true, aF0, true, uF0, accU[0][0], false, false);
    accU[0][1] = __builtin_amdgcn_wmma_i32_16x16x64_iu8(true, aF0, true, uF1, accU[0][1], false, false);
    accU[1][0] = __builtin_amdgcn_wmma_i32_16x16x64_iu8(true, aF1, true, uF0, accU[1][0], false, false);
    accU[1][1] = __builtin_amdgcn_wmma_i32_16x16x64_iu8(true, aF1, true, uF1, accU[1][1], false, false);

    __syncthreads();
  }

  const int rhalf = (lane >> 4) << 3;
  #pragma unroll
  for (int i = 0; i < 2; ++i) {
    #pragma unroll
    for (int r = 0; r < 8; ++r) {
      const int m = bm + wm * 32 + i * 16 + rhalf + r;
      const float xs = xscale[m];
      float amax = 0.f;
      #pragma unroll
      for (int j = 0; j < 2; ++j) {
        const int n = bn + wn * 32 + j * 16 + (lane & 15);
        float g = (float)accG[i][j][r] * xs * wscale[n];
        float u = (float)accU[i][j][r] * xs * wscale[n + IMED];
        float a = g / (1.f + __expf(-g)) * u;
        act[(size_t)m * IMED + n] = a;
        amax = fmaxf(amax, fabsf(a));
      }
      #pragma unroll
      for (int s = 1; s < 16; s <<= 1)
        amax = fmaxf(amax, __shfl_xor(amax, s, 32));
      if ((lane & 15) == 0)
        atomicMax(&s2bits[m], __float_as_uint(amax));
    }
  }
}

__global__ __launch_bounds__(256)
void requant_kernel(const float4* __restrict__ act,
                    const unsigned* __restrict__ s2bits,
                    int* __restrict__ q8, int nquad) {
  int i = blockIdx.x * 256 + threadIdx.x;
  if (i >= nquad) return;
  int row = i / (IMED / 4);
  float mx  = __uint_as_float(s2bits[row]);
  float inv = mx > 0.f ? 127.f / mx : 0.f;
  float4 a = act[i];
  int q0 = q8clamp(a.x * inv);
  int q1 = q8clamp(a.y * inv);
  int q2 = q8clamp(a.z * inv);
  int q3 = q8clamp(a.w * inv);
  q8[i] = q0 | (q1 << 8) | (q2 << 16) | (q3 << 24);
}

__global__ __launch_bounds__(256)
void down_kernel(const char* __restrict__ q8,
                 const char* __restrict__ wd8,
                 const unsigned* __restrict__ s2bits,
                 const float* __restrict__ wdscale,
                 float* __restrict__ out) {
  __shared__ char sA[2][64  * LDS_STRIDE];
  __shared__ char sB[2][128 * LDS_STRIDE];

  const int tid  = threadIdx.x;
  const int lane = tid & 31;
  const int wave = tid >> 5;
  const int wm   = wave & 1;
  const int wn   = wave >> 1;
  const int bm   = blockIdx.x * 64;
  const int bn   = blockIdx.y * 128;

  v8i acc[2][2] = {};

  auto stage = [&](int k0, int b) {
    {
      int row = tid >> 2, ch = (tid & 3) << 4;
      async_b128(lds_off32(&sA[b][row * LDS_STRIDE + ch]), q8,
                 (bm + row) * IMED + k0 + ch);
    }
    #pragma unroll
    for (int s = 0; s < 2; ++s) {
      int t = tid + (s << 8);
      int row = t >> 2, ch = (t & 3) << 4;
      async_b128(lds_off32(&sB[b][row * LDS_STRIDE + ch]), wd8,
                 (bn + row) * IMED + k0 + ch);
    }
  };

  stage(0, 0);
  for (int k0 = 0, it = 0; k0 < IMED; k0 += BK, ++it) {
    const int b = it & 1;
    if (k0 + BK < IMED) {
      stage(k0 + BK, b ^ 1);
      asm volatile("s_wait_asynccnt 3" ::: "memory");
    } else {
      asm volatile("s_wait_asynccnt 0" ::: "memory");
    }
    __syncthreads();

    v8i aF0 = frag_a(sA[b], wm * 32 +  0, lane);
    v8i aF1 = frag_a(sA[b], wm * 32 + 16, lane);
    v8i bF0 = frag_b(sB[b], wn * 32 +  0, lane);
    v8i bF1 = frag_b(sB[b], wn * 32 + 16, lane);

    acc[0][0] = __builtin_amdgcn_wmma_i32_16x16x64_iu8(true, aF0, true, bF0, acc[0][0], false, false);
    acc[0][1] = __builtin_amdgcn_wmma_i32_16x16x64_iu8(true, aF0, true, bF1, acc[0][1], false, false);
    acc[1][0] = __builtin_amdgcn_wmma_i32_16x16x64_iu8(true, aF1, true, bF0, acc[1][0], false, false);
    acc[1][1] = __builtin_amdgcn_wmma_i32_16x16x64_iu8(true, aF1, true, bF1, acc[1][1], false, false);

    __syncthreads();
  }

  const int rhalf = (lane >> 4) << 3;
  #pragma unroll
  for (int i = 0; i < 2; ++i) {
    #pragma unroll
    for (int r = 0; r < 8; ++r) {
      const int m = bm + wm * 32 + i * 16 + rhalf + r;
      const float s2 = __uint_as_float(s2bits[m]) * (1.f / 127.f);
      #pragma unroll
      for (int j = 0; j < 2; ++j) {
        const int n = bn + wn * 32 + j * 16 + (lane & 15);
        out[(size_t)m * HID + n] = (float)acc[i][j][r] * s2 * wdscale[n];
      }
    }
  }
}

extern "C" void kernel_launch(void* const* d_in, const int* in_sizes, int n_in,
                              void* d_out, int out_size, void* d_ws, size_t ws_size,
                              hipStream_t stream) {
  (void)in_sizes; (void)n_in; (void)out_size; (void)ws_size;

  const int*   x_q      = (const int*)d_in[0];
  const float* x_scale  = (const float*)d_in[1];
  const int*   w_gateup = (const int*)d_in[2];
  const float* wg_scale = (const float*)d_in[3];
  const int*   w_down   = (const int*)d_in[4];
  const float* wd_scale = (const float*)d_in[5];
  float*       out      = (float*)d_out;

  char*  ws = (char*)d_ws;
  size_t off = 0;
  auto carve = [&](size_t bytes) {
    char* p = ws + off;
    off = (off + bytes + 255) & ~(size_t)255;
    return p;
  };
  char*     x8   = carve((size_t)N_TOK * HID);
  char*     wg8  = carve((size_t)2 * IMED * HID);
  char*     wd8  = carve((size_t)HID * IMED);
  char*     q2   = carve((size_t)N_TOK * IMED);
  unsigned* s2   = (unsigned*)carve((size_t)N_TOK * 4);
  float*    actf = (float*)carve((size_t)N_TOK * IMED * 4);

  hipMemsetAsync(s2, 0, (size_t)N_TOK * 4, stream);

  const int nx = N_TOK * HID / 4;
  const int ng = 2 * IMED * HID / 4;
  const int nd = HID * IMED / 4;
  pack_i32_to_i8<<<(nx + 255) / 256, 256, 0, stream>>>((const int4*)x_q,      (int*)x8,  nx);
  pack_i32_to_i8<<<(ng + 255) / 256, 256, 0, stream>>>((const int4*)w_gateup, (int*)wg8, ng);
  pack_i32_to_i8<<<(nd + 255) / 256, 256, 0, stream>>>((const int4*)w_down,   (int*)wd8, nd);

  gateup_silu_kernel<<<dim3(N_TOK / 64, IMED / 128), 256, 0, stream>>>(
      x8, wg8, x_scale, wg_scale, actf, s2);

  const int nquad = N_TOK * IMED / 4;
  requant_kernel<<<(nquad + 255) / 256, 256, 0, stream>>>(
      (const float4*)actf, s2, (int*)q2, nquad);

  down_kernel<<<dim3(N_TOK / 64, HID / 128), 256, 0, stream>>>(
      q2, wd8, s2, wd_scale, out);
}
